// GNNLayer_59889023976224
// MI455X (gfx1250) — compile-verified, matrix operand order checked by reading, not run on hardware
//
#include <hip/hip_runtime.h>
#include <hip/hip_fp16.h>

typedef __attribute__((ext_vector_type(16))) _Float16 v16h;
typedef __attribute__((ext_vector_type(8)))  float    v8f;

constexpr int kH   = 128;   // hidden dim
constexpr int kEA  = 16;    // edge attr dim
constexpr int kRows = 64;   // rows (edges/nodes) per block
constexpr int kThreads = 256;

// LDS K-strides (in halves) chosen so b128 fragment loads are bank-conflict-free
constexpr int SW1_E = 168;  // gate-w1 transposed, K padded 144->160 (+8 pad)
constexpr int SW2   = 136;  // 128-K weights (+8 pad)
constexpr int SA_E  = 168;  // edge A = [src|attr|0pad], K=160 used
constexpr int SX1   = 136;  // hidden activation buffer
constexpr int SW1_N = 264;  // node-update w1 transposed, K=256 (+8 pad)
constexpr int SA_N  = 264;  // node A = [x|agg], K=256
constexpr int SU    = 132;  // f32 residual buffer stride (aliases A)

union FragU { v16h v; float4 f[2]; };

// CDNA5 16-bit A/B fragment: lane%16 selects M (A) or N (B); halves hold
// K = {kh..kh+7, kh+16..kh+23}, kh = (lane<16 ? 0 : 8).  Source tile is
// row-major [M_or_N][K] f16 in LDS -> two 16B ds_load_b128 per lane.
__device__ __forceinline__ v16h ld_frag(const _Float16* base, int stride,
                                        int rowBase, int kBase, int lane) {
  int r  = rowBase + (lane & 15);
  int kh = (lane & 16) ? 8 : 0;
  const _Float16* p = base + r * stride + kBase + kh;
  FragU u;
  u.f[0] = *reinterpret_cast<const float4*>(p);
  u.f[1] = *reinterpret_cast<const float4*>(p + 16);
  return u.v;
}

// Fast sigmoid: v_exp_f32 + v_rcp_f32 (avoids the correctly-rounded
// v_div_scale/v_div_fmas sequence the plain divide lowers to).
__device__ __forceinline__ float dev_sigmoid(float x) {
  return __builtin_amdgcn_rcpf(1.0f + __expf(-x));
}

__device__ __forceinline__ void st_half4(_Float16* dst, float a, float b, float c, float d) {
  union { double u; _Float16 h[4]; } t;
  t.h[0] = (_Float16)a; t.h[1] = (_Float16)b; t.h[2] = (_Float16)c; t.h[3] = (_Float16)d;
  *reinterpret_cast<double*>(dst) = t.u;
}

__global__ void zero_f32_kernel(float* __restrict__ p, size_t n4) {
  size_t i = (size_t)blockIdx.x * blockDim.x + threadIdx.x;
  size_t stride = (size_t)gridDim.x * blockDim.x;
  for (; i < n4; i += stride)
    *reinterpret_cast<float4*>(p + i * 4) = float4{0.f, 0.f, 0.f, 0.f};
}

// ---------------------------------------------------------------------------
// Edge phase: msgs = (src@mw + mb) * sigmoid(silu(cat@gw1+gb1)@gw2+gb2),
// atomically scattered into agg[dst].
// Block = 64 edges; wave w owns output columns [16w,16w+16).
// ---------------------------------------------------------------------------
__global__ __launch_bounds__(kThreads)
void edge_msgs_kernel(const float* __restrict__ srcFeat,
                      const int*   __restrict__ edges,
                      const float* __restrict__ eattr,
                      const float* __restrict__ gw1, const float* __restrict__ gb1,
                      const float* __restrict__ gw2, const float* __restrict__ gb2,
                      const float* __restrict__ mw,  const float* __restrict__ mb,
                      float* __restrict__ agg,
                      int E, int srcCol, int dstCol)
{
  extern __shared__ __align__(16) char smem[];
  _Float16* gw1T = (_Float16*)smem;                  // [128][SW1_E]
  _Float16* gw2T = gw1T + 128 * SW1_E;               // [128][SW2]
  _Float16* mwT  = gw2T + 128 * SW2;                 // [128][SW2]
  _Float16* A    = mwT  + 128 * SW2;                 // [64][SA_E]
  _Float16* X1   = A    + kRows * SA_E;              // [64][SX1]
  int*      dIdx = (int*)(X1 + kRows * SX1);         // [64]

  const int tid   = threadIdx.x;
  const int eBase = blockIdx.x * kRows;

  // Stage weights: f32 global -> f16 LDS, transposed to [N][K] (coalesced reads).
  for (int i = tid; i < 128 * 160; i += kThreads) {
    int k = i >> 7, n = i & 127;
    float v = (k < kH + kEA) ? gw1[k * kH + n] : 0.0f;
    gw1T[n * SW1_E + k] = (_Float16)v;
  }
  for (int i = tid; i < 128 * 128; i += kThreads) {
    int k = i >> 7, n = i & 127;
    gw2T[n * SW2 + k] = (_Float16)gw2[k * kH + n];
    mwT [n * SW2 + k] = (_Float16)mw [k * kH + n];
  }

  // Build A = [src(128) | edge_attr(16) | 0-pad(16)] in f16 (4 threads / row).
  {
    int m = tid >> 2, p = tid & 3;
    int e = eBase + m; if (e >= E) e = E - 1;
    int s = edges[e * 2 + srcCol];
    const float* srow = srcFeat + (size_t)s * kH;
    for (int c = p * 32; c < p * 32 + 32; c += 4) {
      float4 v = *reinterpret_cast<const float4*>(srow + c);
      st_half4(&A[m * SA_E + c], v.x, v.y, v.z, v.w);
    }
    float4 va = *reinterpret_cast<const float4*>(eattr + (size_t)e * kEA + p * 4);
    st_half4(&A[m * SA_E + kH + p * 4], va.x, va.y, va.z, va.w);
    st_half4(&A[m * SA_E + kH + kEA + p * 4], 0.f, 0.f, 0.f, 0.f);
    if (p == 0) dIdx[m] = edges[e * 2 + dstCol];
  }
  __syncthreads();

  const int wave = tid >> 5, lane = tid & 31;
  const int nt   = wave;                       // column tile
  const int col  = nt * 16 + (lane & 15);
  const float gb1v = gb1[col];

  // Stage 1: X1 = silu(cat @ gw1 + gb1), K = 160 (5 WMMA steps).
  for (int r = 0; r < 4; ++r) {
    v8f acc = {};
#pragma unroll
    for (int kb = 0; kb < 5; ++kb) {
      v16h a = ld_frag(A,    SA_E,  r * 16, kb * 32, lane);
      v16h b = ld_frag(gw1T, SW1_E, nt * 16, kb * 32, lane);
      acc = __builtin_amdgcn_wmma_f32_16x16x32_f16(false, a, false, b,
                                                   (short)0, acc, false, false);
    }
#pragma unroll
    for (int j = 0; j < 8; ++j) {
      int m = r * 16 + j + ((lane >> 4) << 3);
      float x = acc[j] + gb1v;
      X1[m * SX1 + col] = (_Float16)(x * dev_sigmoid(x));
    }
  }
  __syncthreads();

  // Stage 2: gate = sigmoid(X1@gw2+gb2);  M = src@mw+mb;  scatter M*gate.
  const float gb2v = gb2[col];
  const float mbv  = mb[col];
  for (int r = 0; r < 4; ++r) {
    v8f aG = {}, aM = {};
#pragma unroll
    for (int kb = 0; kb < 4; ++kb) {
      v16h a1 = ld_frag(X1,   SX1,  r * 16, kb * 32, lane);
      v16h b1 = ld_frag(gw2T, SW2,  nt * 16, kb * 32, lane);
      aG = __builtin_amdgcn_wmma_f32_16x16x32_f16(false, a1, false, b1,
                                                  (short)0, aG, false, false);
      v16h a2 = ld_frag(A,    SA_E, r * 16, kb * 32, lane);   // src cols only
      v16h b2 = ld_frag(mwT,  SW2,  nt * 16, kb * 32, lane);
      aM = __builtin_amdgcn_wmma_f32_16x16x32_f16(false, a2, false, b2,
                                                  (short)0, aM, false, false);
    }
#pragma unroll
    for (int j = 0; j < 8; ++j) {
      int m = r * 16 + j + ((lane >> 4) << 3);
      int e = eBase + m;
      if (e < E) {
        float gate = dev_sigmoid(aG[j] + gb2v);
        float msg  = (aM[j] + mbv) * gate;
        atomicAdd(&agg[(size_t)dIdx[m] * kH + col], msg);
      }
    }
  }
}

// ---------------------------------------------------------------------------
// Node phase: out = LayerNorm(x + relu(cat(x,agg)@w1+b1)@w2 + b2).
// Block = 64 nodes; safe when out aliases aggIn (each block only touches
// its own 64 rows, reads complete before writes).
// ---------------------------------------------------------------------------
__global__ __launch_bounds__(kThreads)
void node_update_kernel(const float* __restrict__ X,
                        const float* __restrict__ aggIn,
                        const float* __restrict__ w1, const float* __restrict__ b1,
                        const float* __restrict__ w2, const float* __restrict__ b2,
                        const float* __restrict__ lng, const float* __restrict__ lnb,
                        float* __restrict__ out, int N)
{
  extern __shared__ __align__(16) char smem[];
  _Float16* w1T = (_Float16*)smem;                   // [128][SW1_N]
  _Float16* w2T = w1T + 128 * SW1_N;                 // [128][SW2]
  _Float16* A   = w2T + 128 * SW2;                   // [64][SA_N]
  _Float16* X1  = A + kRows * SA_N;                  // [64][SX1]
  float*    U   = (float*)A;                         // [64][SU] f32, reuses A after stage 1

  const int tid   = threadIdx.x;
  const int nBase = blockIdx.x * kRows;

  for (int i = tid; i < 128 * 256; i += kThreads) {
    int k = i >> 7, n = i & 127;
    w1T[n * SW1_N + k] = (_Float16)w1[k * kH + n];
  }
  for (int i = tid; i < 128 * 128; i += kThreads) {
    int k = i >> 7, n = i & 127;
    w2T[n * SW2 + k] = (_Float16)w2[k * kH + n];
  }
  {
    int m = tid >> 2, p = tid & 3;
    int row = nBase + m; if (row >= N) row = N - 1;
    const float* xr = X     + (size_t)row * kH;
    const float* ar = aggIn + (size_t)row * kH;
    for (int c = p * 32; c < p * 32 + 32; c += 4) {
      float4 v = *reinterpret_cast<const float4*>(xr + c);
      st_half4(&A[m * SA_N + c], v.x, v.y, v.z, v.w);
      float4 g = *reinterpret_cast<const float4*>(ar + c);
      st_half4(&A[m * SA_N + kH + c], g.x, g.y, g.z, g.w);
    }
  }
  __syncthreads();

  const int wave = tid >> 5, lane = tid & 31;
  const int nt   = wave;
  const int col  = nt * 16 + (lane & 15);
  const float b1v = b1[col];

  // Stage 1: X1 = relu(cat @ w1 + b1), K = 256 (8 WMMA steps).
  for (int r = 0; r < 4; ++r) {
    v8f acc = {};
#pragma unroll
    for (int kb = 0; kb < 8; ++kb) {
      v16h a = ld_frag(A,   SA_N,  r * 16, kb * 32, lane);
      v16h b = ld_frag(w1T, SW1_N, nt * 16, kb * 32, lane);
      acc = __builtin_amdgcn_wmma_f32_16x16x32_f16(false, a, false, b,
                                                   (short)0, acc, false, false);
    }
#pragma unroll
    for (int j = 0; j < 8; ++j) {
      int m = r * 16 + j + ((lane >> 4) << 3);
      float x = acc[j] + b1v;
      X1[m * SX1 + col] = (_Float16)fmaxf(x, 0.0f);
    }
  }
  __syncthreads();

  // Stage 2: U = x + X1 @ w2 + b2 (f32, aliases dead A region).
  const float b2v = b2[col];
  for (int r = 0; r < 4; ++r) {
    v8f acc = {};
#pragma unroll
    for (int kb = 0; kb < 4; ++kb) {
      v16h a = ld_frag(X1,  SX1, r * 16, kb * 32, lane);
      v16h b = ld_frag(w2T, SW2, nt * 16, kb * 32, lane);
      acc = __builtin_amdgcn_wmma_f32_16x16x32_f16(false, a, false, b,
                                                   (short)0, acc, false, false);
    }
#pragma unroll
    for (int j = 0; j < 8; ++j) {
      int m = r * 16 + j + ((lane >> 4) << 3);
      int row = nBase + m;
      float xv = (row < N) ? X[(size_t)row * kH + col] : 0.0f;
      U[m * SU + col] = xv + acc[j] + b2v;
    }
  }
  __syncthreads();

  // LayerNorm: wave handles 8 rows; lane covers 4 columns; f32 shfl reductions.
  {
    float4 g4  = *reinterpret_cast<const float4*>(lng + lane * 4);
    float4 bb4 = *reinterpret_cast<const float4*>(lnb + lane * 4);
    for (int mm = wave * 8; mm < wave * 8 + 8; ++mm) {
      float4 v = *reinterpret_cast<const float4*>(&U[mm * SU + lane * 4]);
      float s = v.x + v.y + v.z + v.w;
#pragma unroll
      for (int off = 16; off > 0; off >>= 1) s += __shfl_xor(s, off, 32);
      float mean = s * (1.0f / 128.0f);
      float d0 = v.x - mean, d1 = v.y - mean, d2 = v.z - mean, d3 = v.w - mean;
      float ss = d0 * d0 + d1 * d1 + d2 * d2 + d3 * d3;
#pragma unroll
      for (int off = 16; off > 0; off >>= 1) ss += __shfl_xor(ss, off, 32);
      float rstd = rsqrtf(ss * (1.0f / 128.0f) + 1e-5f);
      int row = nBase + mm;
      if (row < N) {
        float4 o;
        o.x = d0 * rstd * g4.x + bb4.x;
        o.y = d1 * rstd * g4.y + bb4.y;
        o.z = d2 * rstd * g4.z + bb4.z;
        o.w = d3 * rstd * g4.w + bb4.w;
        *reinterpret_cast<float4*>(out + (size_t)row * kH + lane * 4) = o;
      }
    }
  }
}

extern "C" void kernel_launch(void* const* d_in, const int* in_sizes, int n_in,
                              void* d_out, int out_size, void* d_ws, size_t ws_size,
                              hipStream_t stream)
{
  const float* h_hidden = (const float*)d_in[0];
  const float* h_clue   = (const float*)d_in[1];
  const int*   edges    = (const int*)  d_in[2];
  const float* eattr    = (const float*)d_in[3];
  const float* c2h_g_w1 = (const float*)d_in[4];
  const float* c2h_g_b1 = (const float*)d_in[5];
  const float* c2h_g_w2 = (const float*)d_in[6];
  const float* c2h_g_b2 = (const float*)d_in[7];
  const float* c2h_m_w  = (const float*)d_in[8];
  const float* c2h_m_b  = (const float*)d_in[9];
  const float* h_u_w1   = (const float*)d_in[10];
  const float* h_u_b1   = (const float*)d_in[11];
  const float* h_u_w2   = (const float*)d_in[12];
  const float* h_u_b2   = (const float*)d_in[13];
  const float* h_ln_g   = (const float*)d_in[14];
  const float* h_ln_b   = (const float*)d_in[15];
  const float* h2c_g_w1 = (const float*)d_in[16];
  const float* h2c_g_b1 = (const float*)d_in[17];
  const float* h2c_g_w2 = (const float*)d_in[18];
  const float* h2c_g_b2 = (const float*)d_in[19];
  const float* h2c_m_w  = (const float*)d_in[20];
  const float* h2c_m_b  = (const float*)d_in[21];
  const float* c_u_w1   = (const float*)d_in[22];
  const float* c_u_b1   = (const float*)d_in[23];
  const float* c_u_w2   = (const float*)d_in[24];
  const float* c_u_b2   = (const float*)d_in[25];
  const float* c_ln_g   = (const float*)d_in[26];
  const float* c_ln_b   = (const float*)d_in[27];

  const int N = in_sizes[0] / kH;
  const int E = in_sizes[2] / 2;

  float* out  = (float*)d_out;
  float* hNew = out;                        // h_hidden_new -> d_out[0 : N*H)
  float* agg  = out + (size_t)N * kH;       // agg_h, then agg_c, finally h_clue_new

  constexpr size_t SMEM_E =
      (size_t)(128 * SW1_E + 2 * 128 * SW2 + kRows * SA_E + kRows * SX1) * 2 + kRows * 4;
  constexpr size_t SMEM_N =
      (size_t)(128 * SW1_N + 128 * SW2 + kRows * SA_N + kRows * SX1) * 2;

  (void)hipFuncSetAttribute(reinterpret_cast<const void*>(&edge_msgs_kernel),
                            hipFuncAttributeMaxDynamicSharedMemorySize, (int)SMEM_E);
  (void)hipFuncSetAttribute(reinterpret_cast<const void*>(&node_update_kernel),
                            hipFuncAttributeMaxDynamicSharedMemorySize, (int)SMEM_N);

  const size_t NH4 = ((size_t)N * kH) / 4;
  const int eBlocks = (E + kRows - 1) / kRows;
  const int nBlocks = (N + kRows - 1) / kRows;

  // Phase 1: agg_h = segment_sum over h_idx of c2h messages (src = h_clue).
  zero_f32_kernel<<<512, kThreads, 0, stream>>>(agg, NH4);
  edge_msgs_kernel<<<eBlocks, kThreads, SMEM_E, stream>>>(
      h_clue, edges, eattr, c2h_g_w1, c2h_g_b1, c2h_g_w2, c2h_g_b2,
      c2h_m_w, c2h_m_b, agg, E, /*srcCol=*/0, /*dstCol=*/1);
  // Phase 2: h_hidden_new = LN(h_hidden + MLP(cat(h_hidden, agg_h))).
  node_update_kernel<<<nBlocks, kThreads, SMEM_N, stream>>>(
      h_hidden, agg, h_u_w1, h_u_b1, h_u_w2, h_u_b2, h_ln_g, h_ln_b, hNew, N);
  // Phase 3: agg_c = segment_sum over c_idx of h2c messages (src = h_hidden_new).
  zero_f32_kernel<<<512, kThreads, 0, stream>>>(agg, NH4);
  edge_msgs_kernel<<<eBlocks, kThreads, SMEM_E, stream>>>(
      hNew, edges, eattr, h2c_g_w1, h2c_g_b1, h2c_g_w2, h2c_g_b2,
      h2c_m_w, h2c_m_b, agg, E, /*srcCol=*/1, /*dstCol=*/0);
  // Phase 4: h_clue_new written in place over agg_c (per-block row ownership).
  node_update_kernel<<<nBlocks, kThreads, SMEM_N, stream>>>(
      h_clue, agg, c_u_w1, c_u_b1, c_u_w2, c_u_b2, c_ln_g, c_ln_b, agg, N);
}
